// CausalSelfAttention_30769145708613
// MI455X (gfx1250) — matrix operand order repaired and checked by reading, not compiled, not run on hardware
//
#include <hip/hip_runtime.h>
#include <math.h>

#define B_   2
#define T_   2048
#define DIM_ 1024
#define H_   16
#define DH_  64
#define WIN_ 256

typedef __attribute__((ext_vector_type(16))) __bf16 v16bf;
typedef __attribute__((ext_vector_type(8)))  __bf16 v8bf;
typedef __attribute__((ext_vector_type(8)))  float  v8f;

#define WMMA_BF16(a, b, c) \
  __builtin_amdgcn_wmma_f32_16x16x32_bf16(false, (a), false, (b), (short)0, (c), false, false)

// ---- WMMA operand loaders (wave32, 16x16x32 bf16 shapes) ----
// A tile 16x32: row = lane&15; K = 16*(lane>>4) + j
// -> two contiguous 16B halves per lane.
__device__ __forceinline__ v16bf load_A16(const __bf16* __restrict__ src, int ld, int lane) {
  const __bf16* p = src + (size_t)(lane & 15) * ld + ((lane >> 4) << 4);
  v8bf lo = *(const v8bf*)p;
  v8bf hi = *(const v8bf*)(p + 8);
  return __builtin_shufflevector(lo, hi, 0, 1, 2, 3, 4, 5, 6, 7, 8, 9, 10, 11, 12, 13, 14, 15);
}
// B tile 32x16: lane holds col N=lane&15, 16 consecutive K from 16*(lane>>4): one 32B load.
__device__ __forceinline__ v16bf load_B16(const __bf16* __restrict__ p) {
  return *(const v16bf*)p;
}

// ============================================================
// Kernel 0: one-time fp32 -> bf16 conversion (x, wqkv, wout)
// ============================================================
__global__ void cvt_f32_bf16_kernel(const float* __restrict__ s, __bf16* __restrict__ d, int n) {
  const int i = (blockIdx.x * blockDim.x + threadIdx.x) << 2;
  if (i >= n) return;
  const float4 v = *(const float4*)(s + i);
  d[i + 0] = (__bf16)v.x;
  d[i + 1] = (__bf16)v.y;
  d[i + 2] = (__bf16)v.z;
  d[i + 3] = (__bf16)v.w;
}

// ============================================================
// Kernel 1: QKV = X @ Wqkv^T  (bf16 WMMA, fp32 accum)
// one wave computes a 32(M) x 64(N) strip; epilogue scatters to
// Q,K [B,H,T,64] and V transposed [B,H,64,T], all bf16.
// ============================================================
__global__ __launch_bounds__(256) void qkv_gemm_kernel(
    const __bf16* __restrict__ xb, const __bf16* __restrict__ wb,
    __bf16* __restrict__ Q, __bf16* __restrict__ K, __bf16* __restrict__ Vt) {
  const int lane = threadIdx.x & 31;
  const int wave = (blockIdx.x << 3) + (threadIdx.x >> 5);
  const int NSTRIPS = (3 * DIM_) / 64;   // 48
  const int mt = wave / NSTRIPS;
  const int ns = wave - mt * NSTRIPS;
  const int m0 = mt << 5;                // 32 rows in [B*T]
  const int n0 = ns << 6;                // 64 cols in [3*DIM]
  const int nc = lane & 15;
  const int hi = lane >> 4;
  const int kb = hi << 4;

  v8f acc[2][4] = {{{}, {}, {}, {}}, {{}, {}, {}, {}}};
  const __bf16* arow0 = xb + (size_t)m0 * DIM_;
  const __bf16* arow1 = arow0 + (size_t)16 * DIM_;
  for (int kc = 0; kc < DIM_; kc += 32) {
    __builtin_prefetch(arow0 + kc + 128, 0, 0);
    const v16bf a0 = load_A16(arow0 + kc, DIM_, lane);
    const v16bf a1 = load_A16(arow1 + kc, DIM_, lane);
#pragma unroll
    for (int t = 0; t < 4; ++t) {
      const v16bf b = load_B16(wb + (size_t)(n0 + (t << 4) + nc) * DIM_ + kc + kb);
      acc[0][t] = WMMA_BF16(a0, b, acc[0][t]);
      acc[1][t] = WMMA_BF16(a1, b, acc[1][t]);
    }
  }
#pragma unroll
  for (int i = 0; i < 2; ++i) {
    const int mb = m0 + (i << 4) + (hi << 3);
#pragma unroll
    for (int t = 0; t < 4; ++t) {
      const int ncol  = n0 + (t << 4) + nc;
      const int which = ncol >> 10;      // 0=Q 1=K 2=V
      const int rcol  = ncol & 1023;
      const int h     = rcol >> 6;
      const int d     = rcol & 63;
#pragma unroll
      for (int r = 0; r < 8; ++r) {
        const int m  = mb + r;
        const int bb = m >> 11;
        const int tt = m & 2047;
        const __bf16 v = (__bf16)acc[i][t][r];
        const size_t bhh = (size_t)(bb * H_ + h);
        if (which == 0)      Q[(bhh * T_ + tt) * DH_ + d] = v;
        else if (which == 1) K[(bhh * T_ + tt) * DH_ + d] = v;
        else                 Vt[(bhh * DH_ + d) * T_ + tt] = v;  // 8 consecutive tt per lane
      }
    }
  }
}

// ============================================================
// Kernel 2: RoPE in place on bf16 Q, K  (half-split rotation)
// ============================================================
__global__ void rope_kernel(__bf16* __restrict__ Q, __bf16* __restrict__ K) {
  const int idx = blockIdx.x * blockDim.x + threadIdx.x;
  if (idx >= B_ * H_ * T_ * 32) return;
  const int i  = idx & 31;
  const int t  = (idx >> 5) & (T_ - 1);
  const int bh = idx >> 16;
  const float inv = __expf(-0.2878231366242558f * (float)i);  // 10000^(-i/32)
  const float ang = (float)t * inv;
  float sn, cs;
  __sincosf(ang, &sn, &cs);
  const size_t base = ((size_t)bh * T_ + t) * DH_;
  {
    float x1 = (float)Q[base + i], x2 = (float)Q[base + i + 32];
    Q[base + i]      = (__bf16)(x1 * cs - x2 * sn);
    Q[base + i + 32] = (__bf16)(x2 * cs + x1 * sn);
  }
  {
    float x1 = (float)K[base + i], x2 = (float)K[base + i + 32];
    K[base + i]      = (__bf16)(x1 * cs - x2 * sn);
    K[base + i + 32] = (__bf16)(x2 * cs + x1 * sn);
  }
}

// V B-tile loader: fast vector load except in tail chunks near T (wave-uniform branch).
__device__ __forceinline__ v16bf load_B_v(const __bf16* __restrict__ Vrow, int c, int kb16) {
  if (c + 31 < T_) return *(const v16bf*)(Vrow + c + kb16);
  v16bf b;
#pragma unroll
  for (int j = 0; j < 16; ++j) {
    int k = c + kb16 + j;
    if (k > T_ - 1) k = T_ - 1;   // clamped slots are masked (p=0) anyway
    b[j] = Vrow[k];
  }
  return b;
}

// ============================================================
// Kernel 3: windowed flash attention. One wave per 16-query tile.
// ============================================================
__global__ __launch_bounds__(128) void attn_kernel(
    const __bf16* __restrict__ Q, const __bf16* __restrict__ K,
    const __bf16* __restrict__ Vt, __bf16* __restrict__ O) {
  __shared__ __bf16 pbuf[4][16][32];           // per-wave P relayout buffer
  const int lane = threadIdx.x & 31;
  const int w    = threadIdx.x >> 5;
  const int gw   = (blockIdx.x << 2) + w;
  const int qt   = gw & 127;                   // T/16 tiles
  const int bh   = gw >> 7;                    // [0, 32)
  const int qstart = qt << 4;
  const __bf16* Qb = Q  + ((size_t)bh * T_ + qstart) * DH_;
  const __bf16* Kb = K  + (size_t)bh * T_ * DH_;
  const __bf16* Vb = Vt + (size_t)bh * DH_ * T_;

  const v16bf a0 = load_A16(Qb, DH_, lane);       // head dims 0..31
  const v16bf a1 = load_A16(Qb + 32, DH_, lane);  // head dims 32..63

  float Mrow[8], Lrow[8];
  v8f o0 = {}, o1 = {}, o2 = {}, o3 = {};
#pragma unroll
  for (int r = 0; r < 8; ++r) { Mrow[r] = -INFINITY; Lrow[r] = 0.f; }

  const int nc   = lane & 15;
  const int hi   = lane >> 4;
  const int kb16 = hi << 4;
  const float scale = 0.125f;                  // 1/sqrt(64)

  int lo = qstart - (WIN_ - 1); if (lo < 0) lo = 0;
  const int c0 = lo & ~31;

  for (int c = c0; c <= qstart + 15; c += 32) {
    // ---- S = Q K^T (two 16x16 tiles covering 32 keys) ----
    const int key0 = c + nc, key1 = key0 + 16;
    const int ck0 = key0 < T_ ? key0 : T_ - 1;
    const int ck1 = key1 < T_ ? key1 : T_ - 1;
    const v16bf bk00 = load_B16(Kb + (size_t)ck0 * DH_ + kb16);
    const v16bf bk01 = load_B16(Kb + (size_t)ck0 * DH_ + 32 + kb16);
    const v16bf bk10 = load_B16(Kb + (size_t)ck1 * DH_ + kb16);
    const v16bf bk11 = load_B16(Kb + (size_t)ck1 * DH_ + 32 + kb16);
    v8f s0 = {}, s1 = {};
    s0 = WMMA_BF16(a0, bk00, s0);
    s0 = WMMA_BF16(a1, bk01, s0);
    s1 = WMMA_BF16(a0, bk10, s1);
    s1 = WMMA_BF16(a1, bk11, s1);

    // ---- mask + scale + online softmax (row reductions in 16-lane halves) ----
    const int j0 = c + nc, j1 = j0 + 16;
    float px0[8], px1[8], mloc[8];
#pragma unroll
    for (int r = 0; r < 8; ++r) {
      const int q = qstart + (hi << 3) + r;
      const float v0 = ((j0 <= q) && (q - j0 < WIN_)) ? s0[r] * scale : -INFINITY;
      const float v1 = ((j1 <= q) && (q - j1 < WIN_)) ? s1[r] * scale : -INFINITY;
      px0[r] = v0; px1[r] = v1;
      float mx = fmaxf(v0, v1);
      mx = fmaxf(mx, __shfl_xor(mx, 1, 32));
      mx = fmaxf(mx, __shfl_xor(mx, 2, 32));
      mx = fmaxf(mx, __shfl_xor(mx, 4, 32));
      mx = fmaxf(mx, __shfl_xor(mx, 8, 32));
      mloc[r] = mx;
    }
#pragma unroll
    for (int r = 0; r < 8; ++r) {
      const float Mn = fmaxf(Mrow[r], mloc[r]);
      float corr, p0, p1;
      if (Mn == -INFINITY) { corr = 1.f; p0 = 0.f; p1 = 0.f; }
      else {
        corr = __expf(Mrow[r] - Mn);
        p0 = (px0[r] == -INFINITY) ? 0.f : __expf(px0[r] - Mn);
        p1 = (px1[r] == -INFINITY) ? 0.f : __expf(px1[r] - Mn);
      }
      Mrow[r] = Mn;
      float rs = p0 + p1;
      rs += __shfl_xor(rs, 1, 32);
      rs += __shfl_xor(rs, 2, 32);
      rs += __shfl_xor(rs, 4, 32);
      rs += __shfl_xor(rs, 8, 32);
      Lrow[r] = Lrow[r] * corr + rs;
      o0[r] *= corr; o1[r] *= corr; o2[r] *= corr; o3[r] *= corr;
      const int m = (hi << 3) + r;
      pbuf[w][m][nc]      = (__bf16)p0;        // C-layout -> LDS
      pbuf[w][m][nc + 16] = (__bf16)p1;
    }
    __builtin_amdgcn_wave_barrier();
    // ---- reload P in A-layout (16x32): two 16B LDS reads ----
    v16bf pa;
    {
      const __bf16* pp = &pbuf[w][nc][hi << 4];
      v8bf plo = *(const v8bf*)pp;
      v8bf phi = *(const v8bf*)(pp + 8);
      pa = __builtin_shufflevector(plo, phi, 0, 1, 2, 3, 4, 5, 6, 7, 8, 9, 10, 11, 12, 13, 14, 15);
    }
    __builtin_amdgcn_wave_barrier();

    // ---- O += P V  (V transposed: contiguous key loads per lane) ----
    const v16bf bv0 = load_B_v(Vb + (size_t)( 0 + nc) * T_, c, kb16);
    const v16bf bv1 = load_B_v(Vb + (size_t)(16 + nc) * T_, c, kb16);
    const v16bf bv2 = load_B_v(Vb + (size_t)(32 + nc) * T_, c, kb16);
    const v16bf bv3 = load_B_v(Vb + (size_t)(48 + nc) * T_, c, kb16);
    o0 = WMMA_BF16(pa, bv0, o0);
    o1 = WMMA_BF16(pa, bv1, o1);
    o2 = WMMA_BF16(pa, bv2, o2);
    o3 = WMMA_BF16(pa, bv3, o3);
  }

  // ---- normalize + store O as bf16 [B, T, DIM] ----
  const int bb = bh >> 4, h = bh & 15;
#pragma unroll
  for (int r = 0; r < 8; ++r) {
    const float invL = 1.f / Lrow[r];
    const int m = (hi << 3) + r;
    const size_t rowoff = ((size_t)bb * T_ + qstart + m) * DIM_ + h * DH_;
    O[rowoff +  0 + nc] = (__bf16)(o0[r] * invL);
    O[rowoff + 16 + nc] = (__bf16)(o1[r] * invL);
    O[rowoff + 32 + nc] = (__bf16)(o2[r] * invL);
    O[rowoff + 48 + nc] = (__bf16)(o3[r] * invL);
  }
}

// ============================================================
// Kernel 4: Y = O @ Wout^T  (bf16 x bf16 -> fp32 output)
// ============================================================
__global__ __launch_bounds__(256) void out_gemm_kernel(
    const __bf16* __restrict__ O, const __bf16* __restrict__ wb,
    float* __restrict__ Y) {
  const int lane = threadIdx.x & 31;
  const int wave = (blockIdx.x << 3) + (threadIdx.x >> 5);
  const int NSTRIPS = DIM_ / 64;   // 16
  const int mt = wave / NSTRIPS;
  const int ns = wave - mt * NSTRIPS;
  const int m0 = mt << 5;
  const int n0 = ns << 6;
  const int nc = lane & 15;
  const int hi = lane >> 4;
  const int kb = hi << 4;

  v8f acc[2][4] = {{{}, {}, {}, {}}, {{}, {}, {}, {}}};
  const __bf16* arow0 = O + (size_t)m0 * DIM_;
  const __bf16* arow1 = arow0 + (size_t)16 * DIM_;
  for (int kc = 0; kc < DIM_; kc += 32) {
    __builtin_prefetch(arow0 + kc + 128, 0, 0);
    const v16bf a0 = load_A16(arow0 + kc, DIM_, lane);
    const v16bf a1 = load_A16(arow1 + kc, DIM_, lane);
#pragma unroll
    for (int t = 0; t < 4; ++t) {
      const v16bf b = load_B16(wb + (size_t)(n0 + (t << 4) + nc) * DIM_ + kc + kb);
      acc[0][t] = WMMA_BF16(a0, b, acc[0][t]);
      acc[1][t] = WMMA_BF16(a1, b, acc[1][t]);
    }
  }
#pragma unroll
  for (int i = 0; i < 2; ++i) {
    const int mb = m0 + (i << 4) + (hi << 3);
#pragma unroll
    for (int t = 0; t < 4; ++t) {
      const int ncol = n0 + (t << 4) + nc;
#pragma unroll
      for (int r = 0; r < 8; ++r)
        Y[(size_t)(mb + r) * DIM_ + ncol] = acc[i][t][r];
    }
  }
}

extern "C" void kernel_launch(void* const* d_in, const int* in_sizes, int n_in,
                              void* d_out, int out_size, void* d_ws, size_t ws_size,
                              hipStream_t stream) {
  const float* x    = (const float*)d_in[0];   // [2, 2048, 1024]
  const float* wqkv = (const float*)d_in[1];   // [3072, 1024]
  const float* wout = (const float*)d_in[2];   // [1024, 1024]
  float* out        = (float*)d_out;           // [2, 2048, 1024]

  const size_t HS  = (size_t)B_ * H_ * T_ * DH_;  // 4 Mi
  const size_t NX  = (size_t)B_ * T_ * DIM_;      // 4 Mi
  const size_t NW1 = (size_t)3 * DIM_ * DIM_;     // 3 Mi
  const size_t NW2 = (size_t)DIM_ * DIM_;         // 1 Mi

  __bf16* Q   = (__bf16*)d_ws;
  __bf16* K   = Q   + HS;
  __bf16* Vt  = K   + HS;
  __bf16* O   = Vt  + HS;
  __bf16* Xb  = O   + NX;
  __bf16* W1b = Xb  + NX;
  __bf16* W2b = W1b + NW1;                        // total 48 MB of ws

  // 0. one-time fp32 -> bf16 conversion of activations + weights
  cvt_f32_bf16_kernel<<<(int)(NX  >> 10), 256, 0, stream>>>(x,    Xb,  (int)NX);
  cvt_f32_bf16_kernel<<<(int)(NW1 >> 10), 256, 0, stream>>>(wqkv, W1b, (int)NW1);
  cvt_f32_bf16_kernel<<<(int)(NW2 >> 10), 256, 0, stream>>>(wout, W2b, (int)NW2);
  // 1. QKV projection: 128 M-tiles x 48 N-strips = 6144 waves, 8 waves/block
  qkv_gemm_kernel<<<768, 256, 0, stream>>>(Xb, W1b, Q, K, Vt);
  // 2. RoPE on Q, K
  rope_kernel<<<(B_ * H_ * T_ * 32 + 255) / 256, 256, 0, stream>>>(Q, K);
  // 3. Windowed attention: 32 bh x 128 q-tiles = 4096 waves, 4 waves/block
  attn_kernel<<<1024, 128, 0, stream>>>(Q, K, Vt, O);
  // 4. Output projection: 128 M-tiles x 16 N-strips = 2048 waves, 8 waves/block
  out_gemm_kernel<<<256, 256, 0, stream>>>(O, W2b, out);
}
